// MPNNEncoder_32280974197077
// MI455X (gfx1250) — compile-verified, matrix operand order checked by reading, not run on hardware
//
#include <hip/hip_runtime.h>

typedef __attribute__((ext_vector_type(16))) __bf16 v16bf;
typedef __attribute__((ext_vector_type(8)))  float  v8f;

namespace {
constexpr int  kAtomF  = 133;
constexpr int  kBondF  = 14;
constexpr int  kHidden = 300;
constexpr int  kNPad   = 320;   // HIDDEN padded to 20 tiles of 16
constexpr int  kNTiles = 20;
constexpr long kNAtoms = 262144;
constexpr long kNBonds = 524288;
constexpr long kEBB    = 1048576;
constexpr long kNMols  = 16384;
}

// K index held by (lane-half, element j) of a 16-bit WMMA fragment (ISA 7.12.2):
// lanes 0-15: j0..7 -> K0..7,  j8..15 -> K16..23
// lanes16-31: j0..7 -> K8..15, j8..15 -> K24..31
__device__ __forceinline__ int kmap(int half, int j) {
  return (j < 8) ? (j + 8 * half) : (j + 8 + 8 * half);
}

// Pre-swizzle W (row-major [N_out x K_in], used as B = W^T) into fragment-major
// bf16 layout: [kt][nt][lane][j], so GEMM B-fragment = 32 contiguous bytes/lane.
__global__ __launch_bounds__(256) void prep_weight_swizzle(
    const float* __restrict__ W, __bf16* __restrict__ Wb,
    int K_in, int N_out, int KT) {
  int idx = blockIdx.x * 256 + threadIdx.x;
  int total = KT * kNTiles * 32 * 16;
  if (idx >= total) return;
  int j    = idx & 15;
  int lane = (idx >> 4) & 31;
  int nt   = (idx >> 9) % kNTiles;
  int kt   = (idx >> 9) / kNTiles;
  int half = lane >> 4;
  int k = kt * 32 + kmap(half, j);
  int n = nt * 16 + (lane & 15);
  float v = (k < K_in && n < N_out) ? W[(long)n * K_in + k] : 0.0f;
  Wb[idx] = (__bf16)v;
}

// Fused GEMM: OUT[64 x 320 per block] = epi(A @ Wb).
// MODE 0: A = [atom_x[gidx[row]], bond_e[row]]  (K=147 pad 160)
// MODE 1: A = f32 dense, stride kNPad            (K=320)
// MODE 2: A = [atom_x[row], macc[row]]           (K=433 pad 448)
// EPI 0: bf16 relu(acc)   EPI 1: bf16 relu(acc + h0)   EPI 2: f32 relu(acc + bias), col<300
template <int KPAD, int MODE, int EPI>
__global__ __launch_bounds__(256) void gemm_tile(
    const float* __restrict__ xa, const float* __restrict__ xb,
    const int* __restrict__ gidx, const __bf16* __restrict__ Wb,
    const __bf16* __restrict__ h0add, const float* __restrict__ bias,
    __bf16* __restrict__ out_bf, float* __restrict__ out_f) {
  __shared__ __bf16 lds_a[64 * 40];  // 64 rows x 32 K, stride 40 halves (80B rows)
  const int  tid  = threadIdx.x;
  const int  lane = tid & 31;
  const int  wave = tid >> 5;
  const int  mt   = wave & 3;   // 16-row subtile
  const int  ng   = wave >> 2;  // N-tile group (10 tiles each)
  const long row0 = (long)blockIdx.x * 64;

  v8f acc[10] = {};

  constexpr int KT = KPAD / 32;
  for (int kt = 0; kt < KT; ++kt) {
    const int k0 = kt * 32;
    // ---- stage 64x32 A tile into LDS (f32 -> bf16, fused gather/concat) ----
#pragma unroll
    for (int i = 0; i < 8; ++i) {
      int  e   = tid + i * 256;
      int  r   = e >> 5;
      int  kk  = e & 31;
      int  k   = k0 + kk;
      long row = row0 + r;
      float v = 0.0f;
      if (MODE == 0) {
        if (k < kAtomF) {
          int a = gidx[row];
          v = xa[(long)a * kAtomF + k];
        } else if (k < kAtomF + kBondF) {
          v = xb[row * kBondF + (k - kAtomF)];
        }
      } else if (MODE == 1) {
        v = xa[row * kNPad + k];
      } else {
        if (k < kAtomF) v = xa[row * kAtomF + k];
        else            v = xb[row * kNPad + (k - kAtomF)];  // k-133 < 320 always
      }
      lds_a[r * 40 + kk] = (__bf16)v;
    }
    __syncthreads();

    // ---- A fragment: two 16B LDS loads per lane ----
    const int ar   = mt * 16 + (lane & 15);
    const int half = lane >> 4;
    union { v16bf v; uint4 q[2]; } au;
    __builtin_memcpy(&au.q[0], &lds_a[ar * 40 + half * 8], 16);
    __builtin_memcpy(&au.q[1], &lds_a[ar * 40 + 16 + half * 8], 16);

    // ---- B fragments straight from L2-resident swizzled weights ----
    const __bf16* wbase = Wb + (long)kt * (kNTiles * 512) + (long)lane * 16;
#pragma unroll
    for (int t = 0; t < 10; ++t) {
      int nt = ng * 10 + t;
      union { v16bf v; uint4 q[2]; } bu;
      const __bf16* bp = wbase + (long)nt * 512;
      __builtin_memcpy(&bu.q[0], bp, 16);
      __builtin_memcpy(&bu.q[1], bp + 8, 16);
      acc[t] = __builtin_amdgcn_wmma_f32_16x16x32_bf16(
          false, au.v, false, bu.v, (short)0, acc[t], false, false);
    }
    __syncthreads();
  }

  // ---- epilogue: C/D layout M = vgpr + 8*(lane/16), N = lane%16 ----
  const int halfe = lane >> 4;
#pragma unroll
  for (int t = 0; t < 10; ++t) {
    const int nt  = ng * 10 + t;
    const int col = nt * 16 + (lane & 15);
#pragma unroll
    for (int vv = 0; vv < 8; ++vv) {
      const long row = row0 + mt * 16 + vv + 8 * halfe;
      float x = acc[t][vv];
      if (EPI == 0) {
        x = fmaxf(x, 0.0f);
        out_bf[row * kNPad + col] = (__bf16)x;
      } else if (EPI == 1) {
        x += (float)h0add[row * kNPad + col];
        x = fmaxf(x, 0.0f);
        out_bf[row * kNPad + col] = (__bf16)x;
      } else {
        if (col < kHidden) {
          x += bias[col];
          x = fmaxf(x, 0.0f);
          out_f[row * kNPad + col] = x;
        }
      }
    }
  }
}

// segment_sum via f32 atomics: acc[dst[e]] += h[src[e]] (src==nullptr -> identity).
// 75 slots/edge x 4 bf16 columns = 300 real columns exactly.
__global__ __launch_bounds__(256) void scatter_add_kernel(
    const __bf16* __restrict__ h, const int* __restrict__ src,
    const int* __restrict__ dst, float* __restrict__ acc, long n) {
  long t = (long)blockIdx.x * 256 + threadIdx.x;
  long e = t / 75;
  int  slot = (int)(t - e * 75);
  if (e >= n) return;
  int s = src ? src[e] : (int)e;
  int d = dst[e];
  int c = slot * 4;
  struct alignas(8) BF4 { __bf16 x[4]; };
  BF4 hv = *(const BF4*)(h + (long)s * kNPad + c);
  float* ap = acc + (long)d * kNPad + c;
  atomicAdd(ap + 0, (float)hv.x[0]);
  atomicAdd(ap + 1, (float)hv.x[1]);
  atomicAdd(ap + 2, (float)hv.x[2]);
  atomicAdd(ap + 3, (float)hv.x[3]);
}

__global__ __launch_bounds__(256) void pool_scatter_kernel(
    const float* __restrict__ hv, const int* __restrict__ mol_id,
    float* __restrict__ sums, float* __restrict__ counts) {
  long t = (long)blockIdx.x * 256 + threadIdx.x;
  long a = t / 75;
  int  slot = (int)(t - a * 75);
  if (a >= kNAtoms) return;
  int m = mol_id[a];
  int c = slot * 4;
  const float4 v = *(const float4*)(hv + a * kNPad + c);
  float* sp = sums + (long)m * kNPad + c;
  atomicAdd(sp + 0, v.x);
  atomicAdd(sp + 1, v.y);
  atomicAdd(sp + 2, v.z);
  atomicAdd(sp + 3, v.w);
  if (slot == 0) atomicAdd(&counts[m], 1.0f);
}

__global__ __launch_bounds__(256) void pool_div_kernel(
    const float* __restrict__ sums, const float* __restrict__ counts,
    float* __restrict__ out) {
  int t = blockIdx.x * 256 + threadIdx.x;
  if (t >= (int)(kNMols * kHidden)) return;
  int m = t / kHidden;
  int c = t - m * kHidden;
  out[t] = sums[(long)m * kNPad + c] / fmaxf(counts[m], 1.0f);
}

extern "C" void kernel_launch(void* const* d_in, const int* in_sizes, int n_in,
                              void* d_out, int out_size, void* d_ws, size_t ws_size,
                              hipStream_t stream) {
  (void)in_sizes; (void)n_in; (void)out_size; (void)ws_size;
  const float* atom_x        = (const float*)d_in[0];
  const float* bond_e        = (const float*)d_in[1];
  const float* W_i           = (const float*)d_in[2];
  const float* W_h           = (const float*)d_in[3];
  const float* W_o           = (const float*)d_in[4];
  const float* b_o           = (const float*)d_in[5];
  const int*   bond_src_atom = (const int*)d_in[6];
  const int*   bb_src        = (const int*)d_in[7];
  const int*   bb_dst        = (const int*)d_in[8];
  const int*   mol_id        = (const int*)d_in[9];
  float*       out           = (float*)d_out;

  char* ws = (char*)d_ws;
  const size_t szH    = (size_t)kNBonds * kNPad * 2;  // 335.5 MB per bf16 state
  const size_t szMacc = (size_t)kNBonds * kNPad * 4;  // 671 MB f32 accumulator
  __bf16* h0   = (__bf16*)ws;
  __bf16* h    = (__bf16*)(ws + szH);
  float*  macc = (float*)(ws + 2 * szH);
  float*  hv   = macc + (size_t)kNAtoms * kNPad;      // reuse upper half of macc
  char*   wp   = ws + 2 * szH + szMacc;
  __bf16* wi   = (__bf16*)wp;                         // 5  ktiles
  __bf16* wh   = (__bf16*)(wp + (size_t)5 * 10240 * 2);   // 10 ktiles
  __bf16* wo   = (__bf16*)(wp + (size_t)15 * 10240 * 2);  // 14 ktiles
  float*  sums = (float*)(wp + (size_t)29 * 10240 * 2);
  float*  counts = sums + kNMols * kNPad;

  // Swizzle weights into WMMA B-fragment layout (tiny; stays in L2)
  prep_weight_swizzle<<<(5 * 10240 + 255) / 256, 256, 0, stream>>>(W_i, wi, kAtomF + kBondF, kHidden, 5);
  prep_weight_swizzle<<<(10 * 10240 + 255) / 256, 256, 0, stream>>>(W_h, wh, kHidden, kHidden, 10);
  prep_weight_swizzle<<<(14 * 10240 + 255) / 256, 256, 0, stream>>>(W_o, wo, kAtomF + kHidden, kHidden, 14);

  // 1. h0 = relu([x_src, e] @ Wi^T)
  gemm_tile<160, 0, 0><<<(unsigned)(kNBonds / 64), 256, 0, stream>>>(
      atom_x, bond_e, bond_src_atom, wi, nullptr, nullptr, h0, nullptr);

  // 2. two bond->bond message passes: h = relu(h0 + segsum(h[bb_src]) @ Wh^T)
  for (int pass = 0; pass < 2; ++pass) {
    hipMemsetAsync(macc, 0, szMacc, stream);
    const __bf16* hsrc = (pass == 0) ? h0 : h;
    scatter_add_kernel<<<(unsigned)((kEBB * 75) / 256), 256, 0, stream>>>(
        hsrc, bb_src, bb_dst, macc, kEBB);
    gemm_tile<320, 1, 1><<<(unsigned)(kNBonds / 64), 256, 0, stream>>>(
        macc, nullptr, nullptr, wh, h0, nullptr, h, nullptr);
  }

  // 3. bond->atom reduction + h_v = relu([atom_x, m_atom] @ Wo^T + b_o)
  hipMemsetAsync(macc, 0, (size_t)kNAtoms * kNPad * 4, stream);
  scatter_add_kernel<<<(unsigned)((kNBonds * 75) / 256), 256, 0, stream>>>(
      h, nullptr, bond_src_atom, macc, kNBonds);
  gemm_tile<448, 2, 2><<<(unsigned)(kNAtoms / 64), 256, 0, stream>>>(
      atom_x, macc, nullptr, wo, nullptr, b_o, nullptr, hv);

  // 4. mean pool per molecule
  hipMemsetAsync(sums, 0, (size_t)kNMols * kNPad * 4 + (size_t)kNMols * 4, stream);
  pool_scatter_kernel<<<(unsigned)((kNAtoms * 75) / 256), 256, 0, stream>>>(
      hv, mol_id, sums, counts);
  pool_div_kernel<<<(unsigned)((kNMols * kHidden + 255) / 256), 256, 0, stream>>>(
      sums, counts, out);
}
